// EdgePredictionClassifier_4234837754403
// MI455X (gfx1250) — matrix operand order repaired and checked by reading, not compiled, not run on hardware
//
#include <hip/hip_runtime.h>
#include <hip/hip_bf16.h>

typedef __attribute__((ext_vector_type(16))) __bf16 v16bf;
typedef __attribute__((ext_vector_type(8)))  __bf16 v8bf;
typedef __attribute__((ext_vector_type(8)))  float  v8f;

#define D_FEAT 256
#define K_IN   512     // 2*D_FEAT, also K of layer2
#define H1DIM  512
#define H2DIM  256
#define M_TILE 64
#define BLOCK  128

// LDS layout (bytes): h0[64][512]bf16, h1[64][512]bf16, h2[64][256]bf16
#define OFF_H0 0
#define OFF_H1 (M_TILE * K_IN * 2)              // 65536
#define OFF_H2 (OFF_H1 + M_TILE * H1DIM * 2)    // 131072
#define LDS_BYTES (OFF_H2 + M_TILE * H2DIM * 2) // 163840 -> 2 WGs / WGP

// XOR-swizzled LDS addressing: 16-byte chunks permuted by row so that
// 16 consecutive rows at the same column hit 16 distinct bank groups.
__device__ __forceinline__ int swz(int row, int col, int ld) {
    return row * ld + ((((col >> 3) ^ (row & 15)) << 3) | (col & 7));
}

// One 64-wide N-chunk of a 64xK @ KxN GEMM, K = 512, bf16 WMMA, fp32 accum.
// src: LDS (swizzled, ld 512). dst: LDS (swizzled, ld = ldDst). Wb: global bf16.
__device__ __forceinline__ void gemm_chunk64(
    const __bf16* __restrict__ Wb, int ldw,
    const float* __restrict__ bias,
    const __bf16* src, __bf16* dst, int ldDst,
    int n0, int lane)
{
    const int lm    = lane & 15;      // A row within subtile / C column
    const int lh    = lane >> 4;      // K-half selector
    const int koff  = lh * 8;         // A-fragment K offset within 32
    const int laneK = lm + 16 * lh;   // unused; B-fragment K is koff + (s & 7) + 16 * (s >> 3)

    v8f acc[4][4];
#pragma unroll
    for (int m = 0; m < 4; ++m)
#pragma unroll
        for (int n = 0; n < 4; ++n)
#pragma unroll
            for (int i = 0; i < 8; ++i) acc[m][n][i] = 0.0f;

    for (int kk = 0; kk < K_IN; kk += 32) {
        v16bf b[4];
#pragma unroll
        for (int n = 0; n < 4; ++n) {
            const __bf16* bp = Wb + (size_t)(kk + koff) * ldw + (n0 + n * 16 + lm);
#pragma unroll
            for (int s = 0; s < 16; ++s)
                b[n][s] = bp[(size_t)((s & 7) + ((s >> 3) << 4)) * ldw];
        }
#pragma unroll
        for (int m = 0; m < 4; ++m) {
            const int row = m * 16 + lm;
            v8bf alo = *(const v8bf*)(src + swz(row, kk + koff,      512));
            v8bf ahi = *(const v8bf*)(src + swz(row, kk + koff + 16, 512));
            v16bf a = __builtin_shufflevector(alo, ahi,
                                              0,1,2,3,4,5,6,7,8,9,10,11,12,13,14,15);
#pragma unroll
            for (int n = 0; n < 4; ++n)
                acc[m][n] = __builtin_amdgcn_wmma_f32_16x16x32_bf16(
                    false, a, false, b[n], (short)0, acc[m][n], false, false);
        }
    }

    // bias + ReLU + bf16 store (C layout: VGPR i, lane l -> M = i + 8*(l>>4), N = l&15)
#pragma unroll
    for (int n = 0; n < 4; ++n) {
        const int ncol = n0 + n * 16 + lm;
        const float bv = bias[ncol];
#pragma unroll
        for (int m = 0; m < 4; ++m) {
#pragma unroll
            for (int i = 0; i < 8; ++i) {
                float v = acc[m][n][i] + bv;
                v = v > 0.0f ? v : 0.0f;
                dst[swz(m * 16 + i + 8 * lh, ncol, ldDst)] = (__bf16)v;
            }
        }
    }
}

__global__ void convert_weights_kernel(const float* __restrict__ W0,
                                       const float* __restrict__ W1,
                                       __bf16* __restrict__ W0b,
                                       __bf16* __restrict__ W1b)
{
    const int n0 = 512 * 512, n1 = 512 * 256;
    for (int i = blockIdx.x * blockDim.x + threadIdx.x; i < n0 + n1;
         i += gridDim.x * blockDim.x) {
        if (i < n0) W0b[i] = (__bf16)W0[i];
        else        W1b[i - n0] = (__bf16)W1[i - n0];
    }
}

__global__ void __launch_bounds__(BLOCK) edge_mlp_kernel(
    const float* __restrict__ x,
    const int*   __restrict__ eiA,
    const int*   __restrict__ eiB,
    const __bf16* __restrict__ W0b, const float* __restrict__ b0,
    const __bf16* __restrict__ W1b, const float* __restrict__ b1,
    const float* __restrict__ W2,   const float* __restrict__ b2,
    float* __restrict__ out, int E)
{
    extern __shared__ char smem[];
    __bf16* h0 = (__bf16*)(smem + OFF_H0);
    __bf16* h1 = (__bf16*)(smem + OFF_H1);
    __bf16* h2 = (__bf16*)(smem + OFF_H2);

    const int tid      = threadIdx.x;
    const int set      = blockIdx.y;
    const int tileBase = blockIdx.x * M_TILE;
    const int* ei      = set ? eiB : eiA;

    // ---- gather + concat + fp32->bf16 into h0 (64 rows x 512 cols) ----
    for (int c = tid; c < M_TILE * 64; c += BLOCK) {  // 64 x 8-elem chunks per row
        const int r = c >> 6;
        const int p = c & 63;
        const int e = tileBase + r;
        __bf16* dstp = h0 + swz(r, p * 8, 512);
        if (e < E) {
            const int node = (p < 32) ? ei[e] : ei[E + e];
            const int colb = (p & 31) * 8;
            const float4* sp = (const float4*)(x + (size_t)node * D_FEAT + colb);
            const float4 f0 = sp[0], f1 = sp[1];
            v8bf o;
            o[0] = (__bf16)f0.x; o[1] = (__bf16)f0.y;
            o[2] = (__bf16)f0.z; o[3] = (__bf16)f0.w;
            o[4] = (__bf16)f1.x; o[5] = (__bf16)f1.y;
            o[6] = (__bf16)f1.z; o[7] = (__bf16)f1.w;
            *(v8bf*)dstp = o;
        } else {
            v8bf z;
#pragma unroll
            for (int j = 0; j < 8; ++j) z[j] = (__bf16)0.0f;
            *(v8bf*)dstp = z;
        }
    }
    __syncthreads();

    const int wave = tid >> 5;
    const int lane = tid & 31;

    // ---- layer 1: h0[64x512] @ W0[512x512] -> relu -> h1 (wave owns 128 cols) ----
    gemm_chunk64(W0b, H1DIM, b0, h0, h1, H1DIM, wave * 128,      lane);
    gemm_chunk64(W0b, H1DIM, b0, h0, h1, H1DIM, wave * 128 + 64, lane);
    __syncthreads();

    // ---- layer 2: h1[64x512] @ W1[512x256] -> relu -> h2 (wave owns 64 cols) ----
    gemm_chunk64(W1b, H2DIM, b1, h1, h2, H2DIM, wave * 64, lane);
    __syncthreads();

    // ---- layer 3 (N=2) + log_softmax, fp32 VALU ----
    {
        const int el  = tid >> 1;   // local edge
        const int cls = tid & 1;    // class
        float s = 0.0f;
#pragma unroll
        for (int c = 0; c < H2DIM / 8; ++c) {
            const v8bf hv = *(const v8bf*)(h2 + swz(el, c * 8, 256));
#pragma unroll
            for (int j = 0; j < 8; ++j)
                s += (float)hv[j] * W2[(c * 8 + j) * 2 + cls];
        }
        s += b2[cls];
        const float o2  = __shfl_xor(s, 1, 32);
        const float mx  = fmaxf(s, o2);
        const float lse = mx + __logf(__expf(s - mx) + __expf(o2 - mx));
        const int e = tileBase + el;
        if (e < E)
            out[(size_t)set * (size_t)E * 2 + (size_t)e * 2 + cls] = s - lse;
    }
}

extern "C" void kernel_launch(void* const* d_in, const int* in_sizes, int n_in,
                              void* d_out, int out_size, void* d_ws, size_t ws_size,
                              hipStream_t stream) {
    const float* x   = (const float*)d_in[0];
    const int*   eiA = (const int*)d_in[1];
    const int*   eiB = (const int*)d_in[2];
    const float* W0  = (const float*)d_in[3];
    const float* b0  = (const float*)d_in[4];
    const float* W1  = (const float*)d_in[5];
    const float* b1  = (const float*)d_in[6];
    const float* W2  = (const float*)d_in[7];
    const float* b2  = (const float*)d_in[8];
    float* out = (float*)d_out;

    const int E = in_sizes[1] / 2;   // edge_index is (2, E)

    __bf16* W0b = (__bf16*)d_ws;                 // 512*512 bf16 = 512 KB
    __bf16* W1b = W0b + 512 * 512;               // 512*256 bf16 = 256 KB

    convert_weights_kernel<<<384, 256, 0, stream>>>(W0, W1, W0b, W1b);

    (void)hipFuncSetAttribute((const void*)edge_mlp_kernel,
                              hipFuncAttributeMaxDynamicSharedMemorySize,
                              LDS_BYTES);

    dim3 grid((E + M_TILE - 1) / M_TILE, 2);
    edge_mlp_kernel<<<grid, BLOCK, LDS_BYTES, stream>>>(
        x, eiA, eiB, W0b, b0, W1b, b1, W2, b2, out, E);
}
